// GCNModel_89970974917472
// MI455X (gfx1250) — matrix operand order repaired and checked by reading, not compiled, not run on hardware
//
#include <hip/hip_runtime.h>
#include <hip/hip_bf16.h>
#include <stdint.h>

#define NN 100000
#define NE 1600000
#define CH 128
#define NG 512
#define NPAD 100096   // 782 * 128

typedef __attribute__((ext_vector_type(16))) __bf16 v16bf;
typedef __attribute__((ext_vector_type(8)))  float  v8f;
typedef __attribute__((ext_vector_type(4)))  unsigned int v4u;
typedef __attribute__((ext_vector_type(8)))  unsigned int v8u;

union Frag { v16bf v; uint4 q[2]; };

__device__ __forceinline__ unsigned short f2bf(float f) {
  unsigned u = __float_as_uint(f);
  unsigned r = u + 0x7FFFu + ((u >> 16) & 1u);   // round-to-nearest-even
  return (unsigned short)(r >> 16);
}

// ---------------- utility fills / converts ----------------
__global__ void fill_f32(float* p, float v, int n) {
  int i = blockIdx.x * blockDim.x + threadIdx.x;
  if (i < n) p[i] = v;
}
__global__ void fill_i32(int* p, int v, int n) {
  int i = blockIdx.x * blockDim.x + threadIdx.x;
  if (i < n) p[i] = v;
}
__global__ void cvt_bf16(const float* __restrict__ s, unsigned short* __restrict__ d,
                         int nvalid, int ntot) {
  int i = blockIdx.x * blockDim.x + threadIdx.x;
  if (i < ntot) d[i] = (i < nvalid) ? f2bf(s[i]) : (unsigned short)0;
}

// ---------------- degree / norm ----------------
__global__ void deg_accum(const int* __restrict__ dst, float* __restrict__ deg, int E) {
  int e = blockIdx.x * blockDim.x + threadIdx.x;
  if (e < E)
    __hip_atomic_fetch_add(&deg[dst[e]], 1.0f, __ATOMIC_RELAXED, __HIP_MEMORY_SCOPE_AGENT);
}
__global__ void deg_to_dinv(float* d, int n) {
  int i = blockIdx.x * blockDim.x + threadIdx.x;
  if (i < n) { float v = d[i]; d[i] = (v > 0.f) ? rsqrtf(v) : 0.f; }
}

// ---------------- WMMA bf16 GEMM: C[NPAD,128] = A[NPAD,128] x W[128,128] ----------------
// A tile (128x128 bf16, contiguous 32KB) staged via Tensor Data Mover;
// W staged manually with transpose so B-fragments are K-contiguous.
__global__ __launch_bounds__(256)
void gemm_wmma_bf16(const unsigned short* __restrict__ A,
                    const unsigned short* __restrict__ Wg,
                    float* __restrict__ C, int nblocks) {
  __shared__ unsigned short As[128 * 128];   // 32 KB
  __shared__ unsigned short Wt[128 * 128];   // 32 KB (transposed: Wt[n][k])
  const int t = threadIdx.x;
  const int mblk = blockIdx.x;

  // ---- TDM: DMA the A tile into LDS (wave 0 issues; TDM ignores EXEC) ----
  if ((t >> 5) == 0) {
    const unsigned long long ga =
        (unsigned long long)(const void*)(A + (size_t)mblk * 128 * 128);
    const unsigned lds = (unsigned)(unsigned long long)(const void*)&As[0];
    // D# group 0: count=1 | lds_addr | global_addr[56:0] | type=2
    v4u g0;
    g0.x = 1u;                                        // count=1, is_restore=0
    g0.y = lds;                                       // lds_addr
    g0.z = (unsigned)ga;                              // global_addr[31:0]
    g0.w = ((unsigned)(ga >> 32) & 0x01FFFFFFu) | 0x80000000u;  // [56:32] | type=2
    // D# group 1: data_size=1 (2B); tensor 128x128; tile 128x128; stride0=128
    v8u g1;
    g1.s0 = 0x10000u;          // workgroup_mask=0, data_size=1 (2 bytes)
    g1.s1 = 128u << 16;        // tensor_dim0 = 128  (bits 79:48)
    g1.s2 = 128u << 16;        // tensor_dim1 = 128  (bits 111:80)
    g1.s3 = 128u << 16;        // tile_dim0  = 128   (bits 127:112)
    g1.s4 = 128u;              // tile_dim1  = 128   (bits 143:128)
    g1.s5 = 128u;              // tensor_dim0_stride = 128 (bits 207:160)
    g1.s6 = 0u;
    g1.s7 = 0u;
    const v4u gz = {0u, 0u, 0u, 0u};   // groups 2/3: tile_dim3/4 = 0 -> unused (2D)
    asm volatile("tensor_load_to_lds %0, %1, %2, %3"
                 :: "s"(g0), "s"(g1), "s"(gz), "s"(gz)
                 : "memory");
  }

  // Prefetch next block's A tile (gfx1250 global_prefetch_b8).
  if (mblk + 1 < nblocks)
    __builtin_prefetch(A + (size_t)(mblk + 1) * 128 * 128, 0, 1);

  // Stage W with transpose while the TDM DMA is in flight.
#pragma unroll 4
  for (int i = 0; i < 64; ++i) {
    int lin = t + 256 * i;           // 0..16383
    int k = lin >> 7, n = lin & 127;
    Wt[n * 128 + k] = Wg[lin];
  }

  if ((t >> 5) == 0) __builtin_amdgcn_s_wait_tensorcnt(0);
  __syncthreads();

  const int w  = t >> 5;        // wave 0..7 -> M strip
  const int l  = t & 31;
  const int g  = l >> 4;        // lane half
  const int ll = l & 15;
  const int mw = w * 16;

  const v8f vzero = {0.f, 0.f, 0.f, 0.f, 0.f, 0.f, 0.f, 0.f};
  v8f acc[8];
#pragma unroll
  for (int i = 0; i < 8; ++i) acc[i] = vzero;

#pragma unroll
  for (int kc4 = 0; kc4 < 4; ++kc4) {
    const int kc = kc4 * 32;
    // A fragment: lane ll holds row mw+ll; K-halves split by lane group.
    Frag a;
    const uint4* ap = (const uint4*)&As[(mw + ll) * 128 + kc + 16 * g];
    a.q[0] = ap[0];          // K = kc+16g .. +7
    a.q[1] = ap[1];          // K = kc+16g+8 .. +15
#pragma unroll
    for (int nt = 0; nt < 8; ++nt) {
      Frag b;
      const uint4* bp = (const uint4*)&Wt[(nt * 16 + ll) * 128 + kc + 16 * g];
      b.q[0] = bp[0];        // K = kc+16g .. +7
      b.q[1] = bp[1];        // K = kc+16g+8 .. +15
      acc[nt] = __builtin_amdgcn_wmma_f32_16x16x32_bf16(
          false, a.v, false, b.v, (short)0, acc[nt], false, false);
    }
  }

  // Store: VGPR j of C/D tile is row M = j + 8*g.
  float* Cb = C + (size_t)mblk * 128 * 128;
#pragma unroll
  for (int nt = 0; nt < 8; ++nt) {
    const int col = nt * 16 + ll;
#pragma unroll
    for (int j = 0; j < 8; ++j) {
      const int row = mw + j + 8 * g;
      Cb[row * 128 + col] = acc[nt][j];
    }
  }
}

// ---------------- edge gather/scale/scatter (one wave per edge) ----------------
__global__ __launch_bounds__(256)
void edge_scatter(const int* __restrict__ src, const int* __restrict__ dst,
                  const float* __restrict__ dinv, const float* __restrict__ t,
                  float* __restrict__ agg, int E) {
  int e = blockIdx.x * 8 + (threadIdx.x >> 5);
  if (e >= E) return;
  const int lane = threadIdx.x & 31;
  const int s = src[e], d = dst[e];
  const float wgt = dinv[s] * dinv[d];
  const float4 v = ((const float4*)(t + (size_t)s * CH))[lane];
  float* ad = agg + (size_t)d * CH + lane * 4;
  __hip_atomic_fetch_add(ad + 0, v.x * wgt, __ATOMIC_RELAXED, __HIP_MEMORY_SCOPE_AGENT);
  __hip_atomic_fetch_add(ad + 1, v.y * wgt, __ATOMIC_RELAXED, __HIP_MEMORY_SCOPE_AGENT);
  __hip_atomic_fetch_add(ad + 2, v.z * wgt, __ATOMIC_RELAXED, __HIP_MEMORY_SCOPE_AGENT);
  __hip_atomic_fetch_add(ad + 3, v.w * wgt, __ATOMIC_RELAXED, __HIP_MEMORY_SCOPE_AGENT);
}

// ---------------- self-loop + bias + relu + residual ----------------
__global__ void combine(const float* __restrict__ agg, const float* __restrict__ t,
                        const float* __restrict__ dinv, const float* __restrict__ bias,
                        const float* __restrict__ hres, float* __restrict__ hout, int total) {
  int i = blockIdx.x * blockDim.x + threadIdx.x;
  if (i >= total) return;
  const int n = i >> 7, c = i & 127;
  const float di = dinv[n];
  float v = agg[i] + di * di * t[i] + bias[c];
  v = fmaxf(v, 0.f);
  if (hres) v += hres[i];
  hout[i] = v;
}

// ---------------- pooling (mean + max per graph) ----------------
__global__ void pool_accum(const float* __restrict__ h, const int* __restrict__ batch,
                           float* __restrict__ gsum, int* __restrict__ gmaxb,
                           float* __restrict__ cnt) {
  const int n = blockIdx.x, c = threadIdx.x;
  const int g = batch[n];
  const float v = h[(size_t)n * CH + c];
  __hip_atomic_fetch_add(&gsum[g * CH + c], v, __ATOMIC_RELAXED, __HIP_MEMORY_SCOPE_AGENT);
  atomicMax(&gmaxb[g * CH + c], __float_as_int(v));   // h >= 0 -> int-order == float-order
  if (c == 0)
    __hip_atomic_fetch_add(&cnt[g], 1.0f, __ATOMIC_RELAXED, __HIP_MEMORY_SCOPE_AGENT);
}

// ---------------- MLP head ----------------
__global__ void head1(const float* __restrict__ gsum, const int* __restrict__ gmaxb,
                      const float* __restrict__ cnt, const float* __restrict__ Wf1,
                      const float* __restrict__ bf1, float* __restrict__ ghid) {
  const int g = blockIdx.x, j = threadIdx.x;   // 512 x 128
  const float inv = 1.0f / fmaxf(cnt[g], 1.0f);
  float s = bf1[j];
  for (int k = 0; k < CH; ++k) s += gsum[g * CH + k] * inv * Wf1[k * CH + j];
  for (int k = 0; k < CH; ++k) s += __int_as_float(gmaxb[g * CH + k]) * Wf1[(CH + k) * CH + j];
  ghid[g * CH + j] = fmaxf(s, 0.f);
}
__global__ void head2(const float* __restrict__ ghid, const float* __restrict__ Wf2,
                      const float* __restrict__ bf2, float* __restrict__ gh2) {
  const int g = blockIdx.x, j = threadIdx.x;   // 512 x 64
  float s = bf2[j];
  for (int k = 0; k < CH; ++k) s += ghid[g * CH + k] * Wf2[k * 64 + j];
  gh2[g * 64 + j] = fmaxf(s, 0.f);
}
__global__ void head3(const float* __restrict__ gh2, const float* __restrict__ Wf3,
                      const float* __restrict__ bf3, float* __restrict__ out) {
  int g = blockIdx.x * blockDim.x + threadIdx.x;
  if (g >= NG) return;
  float s = bf3[0];
  for (int k = 0; k < 64; ++k) s += gh2[g * 64 + k] * Wf3[k];
  out[g] = s;
}

extern "C" void kernel_launch(void* const* d_in, const int* in_sizes, int n_in,
                              void* d_out, int out_size, void* d_ws, size_t ws_size,
                              hipStream_t stream) {
  (void)in_sizes; (void)n_in; (void)out_size; (void)ws_size;
  const float* x    = (const float*)d_in[0];
  const int*   eidx = (const int*)d_in[1];
  const int*   batch= (const int*)d_in[2];
  const float* Wl[3] = {(const float*)d_in[4], (const float*)d_in[6], (const float*)d_in[8]};
  const float* bl[3] = {(const float*)d_in[5], (const float*)d_in[7], (const float*)d_in[9]};
  const float* Wf1 = (const float*)d_in[10]; const float* bf1 = (const float*)d_in[11];
  const float* Wf2 = (const float*)d_in[12]; const float* bf2 = (const float*)d_in[13];
  const float* Wf3 = (const float*)d_in[14]; const float* bf3 = (const float*)d_in[15];
  float* out = (float*)d_out;

  char* ws = (char*)d_ws;
  size_t off = 0;
  auto take = [&](size_t bytes) -> char* {
    char* p = ws + off; off += (bytes + 255) & ~(size_t)255; return p;
  };
  float* dinv  = (float*)take((size_t)NPAD * 4);
  float* h     = (float*)take((size_t)NPAD * CH * 4);
  float* tbuf  = (float*)take((size_t)NPAD * CH * 4);
  float* agg   = (float*)take((size_t)NPAD * CH * 4);
  unsigned short* hb  = (unsigned short*)take((size_t)NPAD * CH * 2);
  unsigned short* Wb[3];
  Wb[0] = (unsigned short*)take(128 * 128 * 2);
  Wb[1] = (unsigned short*)take(128 * 128 * 2);
  Wb[2] = (unsigned short*)take(128 * 128 * 2);
  float* gsum  = (float*)take((size_t)NG * CH * 4);
  int*   gmaxb = (int*)take((size_t)NG * CH * 4);
  float* cnt   = (float*)take((size_t)NG * 4);
  float* ghid  = (float*)take((size_t)NG * CH * 4);
  float* gh2   = (float*)take((size_t)NG * 64 * 4);

  const int TPB = 256;
  const int nblocks = NPAD / 128;                 // 782
  const int ncElems = NN * CH;                    // 12.8M valid
  const int npElems = NPAD * CH;                  // padded

  // degrees (self-loop folded in as init=1)
  fill_f32<<<(NN + TPB - 1) / TPB, TPB, 0, stream>>>(dinv, 1.0f, NN);
  deg_accum<<<(NE + TPB - 1) / TPB, TPB, 0, stream>>>(eidx + NE, dinv, NE);
  deg_to_dinv<<<(NN + TPB - 1) / TPB, TPB, 0, stream>>>(dinv, NN);

  // weights -> bf16
  for (int i = 0; i < 3; ++i)
    cvt_bf16<<<(128 * 128 + TPB - 1) / TPB, TPB, 0, stream>>>(Wl[i], Wb[i], 128 * 128, 128 * 128);

  // 3 GCN layers
  for (int layer = 0; layer < 3; ++layer) {
    const float* srcH = (layer == 0) ? x : h;
    cvt_bf16<<<(npElems + TPB - 1) / TPB, TPB, 0, stream>>>(srcH, hb, ncElems, npElems);
    gemm_wmma_bf16<<<nblocks, 256, 0, stream>>>(hb, Wb[layer], tbuf, nblocks);
    fill_f32<<<(ncElems + TPB - 1) / TPB, TPB, 0, stream>>>(agg, 0.f, ncElems);
    edge_scatter<<<(NE + 7) / 8, 256, 0, stream>>>(eidx, eidx + NE, dinv, tbuf, agg, NE);
    combine<<<(ncElems + TPB - 1) / TPB, TPB, 0, stream>>>(
        agg, tbuf, dinv, bl[layer], (layer == 0) ? (const float*)nullptr : h, h, ncElems);
  }

  // pooling
  fill_f32<<<(NG * CH + TPB - 1) / TPB, TPB, 0, stream>>>(gsum, 0.f, NG * CH);
  fill_i32<<<(NG * CH + TPB - 1) / TPB, TPB, 0, stream>>>(gmaxb, 0, NG * CH);
  fill_f32<<<(NG + TPB - 1) / TPB, TPB, 0, stream>>>(cnt, 0.f, NG);
  pool_accum<<<NN, CH, 0, stream>>>(h, batch, gsum, gmaxb, cnt);

  // head
  head1<<<NG, CH, 0, stream>>>(gsum, gmaxb, cnt, Wf1, bf1, ghid);
  head2<<<NG, 64, 0, stream>>>(ghid, Wf2, bf2, gh2);
  head3<<<(NG + TPB - 1) / TPB, TPB, 0, stream>>>(gh2, Wf3, bf3, out);
}
